// NeWCRFBlock_40681930227906
// MI455X (gfx1250) — matrix operand order repaired and checked by reading, not compiled, not run on hardware
//
#include <hip/hip_runtime.h>
#include <math.h>

// ---------------------------------------------------------------------------
// Types & helpers
// ---------------------------------------------------------------------------
typedef __attribute__((ext_vector_type(16))) __bf16 bf16x16;
typedef __attribute__((ext_vector_type(8)))  __bf16 bf16x8;
typedef __attribute__((ext_vector_type(8)))  float  f32x8;

#define DEV __device__ __forceinline__

DEV __bf16 f2bf(float f) {
  union { float f; unsigned u; } a; a.f = f;
  unsigned r = a.u + 0x7FFFu + ((a.u >> 16) & 1u);   // round-to-nearest-even
  unsigned short h = (unsigned short)(r >> 16);
  return __builtin_bit_cast(__bf16, h);
}

DEV float gelu_f(float x) {            // exact GELU (matches approximate=False)
  return 0.5f * x * (1.0f + erff(x * 0.70710678118654752f));
}

// raw cross-lane gather (wave-internal, no bounds-check scaffolding)
DEV float bperm_f32(int byteaddr, float v) {
  return __builtin_bit_cast(
      float, __builtin_amdgcn_ds_bpermute(byteaddr, __builtin_bit_cast(int, v)));
}

// A-operand fragment (16x32 bf16): lane r=lane%16, half=lane/16
// holds K = half*16 + [0..15] contiguous.
DEV bf16x16 load_fragA(const __bf16* p, int ld, int r, int half) {
  const __bf16* q = p + (size_t)r * ld + half * 16;
  bf16x8 lo = *(const bf16x8*)(q);
  bf16x8 hi = *(const bf16x8*)(q + 8);
  return __builtin_shufflevector(lo, hi, 0,1,2,3,4,5,6,7,8,9,10,11,12,13,14,15);
}

// B-operand fragment (32x16 bf16) read from an N x K (column-major of B) array:
// lane holds column n = lane%16, K = half*16 + [0..15] contiguous.
DEV bf16x16 load_fragB(const __bf16* p, int ld, int r, int half) {
  return *(const bf16x16*)(p + (size_t)r * ld + half * 16);
}

DEV f32x8 wmma_bf16(bf16x16 a, bf16x16 b, f32x8 c) {
  return __builtin_amdgcn_wmma_f32_16x16x32_bf16(false, a, false, b, (short)0, c,
                                                 false, false);
}

// window-row -> original token index (window reverse permutation)
DEV long win_reverse_token(long wr) {
  long win = wr >> 6; int t = (int)(wr & 63);
  long b  = win >> 8;
  int  wh = (int)((win >> 4) & 15);
  int  ww = (int)(win & 15);
  int  h  = wh * 8 + (t >> 3);
  int  w  = ww * 8 + (t & 7);
  return b * 16384 + (long)h * 128 + w;
}

// ---------------------------------------------------------------------------
// Small prep kernels
// ---------------------------------------------------------------------------
__global__ void cvt_bf16_kernel(const float* __restrict__ in,
                                __bf16* __restrict__ out, long n) {
  long i = (long)blockIdx.x * blockDim.x + threadIdx.x;
  if (i < n) out[i] = f2bf(in[i]);
}

// W: K x N (fp32, row-major)  ->  Wt: N x K (bf16, row-major)
__global__ __launch_bounds__(256)
void transpose_w_kernel(const float* __restrict__ W, __bf16* __restrict__ Wt,
                        int K, int N) {
  __shared__ float tile[32][33];
  int n0 = blockIdx.x * 32, k0 = blockIdx.y * 32;
  int tx = threadIdx.x, ty = threadIdx.y;
  for (int j = 0; j < 4; ++j) {
    int k = k0 + ty + 8 * j;
    tile[ty + 8 * j][tx] = W[(size_t)k * N + n0 + tx];
  }
  __syncthreads();
  for (int j = 0; j < 4; ++j) {
    int n = n0 + ty + 8 * j;
    Wt[(size_t)n * K + k0 + tx] = f2bf(tile[tx][ty + 8 * j]);
  }
}

// feature [B=4][C=512][H=128][W=128] -> fwin [1024 windows * 64 tokens][512]
__global__ __launch_bounds__(256)
void partition_kernel(const float* __restrict__ feat, float* __restrict__ fwin) {
  __shared__ float tile[32][33];
  int b = blockIdx.z >> 7, h = blockIdx.z & 127;
  int w0 = blockIdx.x * 32, c0 = blockIdx.y * 32;
  int tx = threadIdx.x, ty = threadIdx.y;
  for (int j = 0; j < 4; ++j) {
    int c = c0 + ty + 8 * j;
    tile[ty + 8 * j][tx] = feat[(((size_t)b * 512 + c) * 128 + h) * 128 + w0 + tx];
  }
  __syncthreads();
  int wh = h >> 3, ih = h & 7;
  for (int j = 0; j < 4; ++j) {
    int w = w0 + ty + 8 * j;
    int ww = w >> 3, iw = w & 7;
    long wrow = ((((long)b * 16 + wh) * 16 + ww) << 6) + ih * 8 + iw;
    fwin[wrow * 512 + c0 + tx] = tile[tx][ty + 8 * j];
  }
}

// one 128-thread block per 512-wide row; fp32 stats, bf16 output
__global__ __launch_bounds__(128)
void layernorm_kernel(const float* __restrict__ in, const float* __restrict__ g,
                      const float* __restrict__ b, __bf16* __restrict__ out) {
  __shared__ float red[128];
  long row = blockIdx.x;
  const float* p = in + row * 512;
  int t = threadIdx.x;
  float x[4]; float s = 0.f;
  for (int i = 0; i < 4; ++i) { x[i] = p[t + 128 * i]; s += x[i]; }
  red[t] = s; __syncthreads();
  for (int off = 64; off > 0; off >>= 1) { if (t < off) red[t] += red[t + off]; __syncthreads(); }
  float mean = red[0] * (1.f / 512.f);
  __syncthreads();
  float vs = 0.f;
  for (int i = 0; i < 4; ++i) { float d = x[i] - mean; vs += d * d; }
  red[t] = vs; __syncthreads();
  for (int off = 64; off > 0; off >>= 1) { if (t < off) red[t] += red[t + off]; __syncthreads(); }
  float inv = rsqrtf(red[0] * (1.f / 512.f) + 1e-5f);
  for (int i = 0; i < 4; ++i) {
    int c = t + 128 * i;
    out[row * 512 + c] = f2bf((x[i] - mean) * inv * g[c] + b[c]);
  }
}

// bias_table[225][16] -> biasT[16][64][64]
__global__ void bias_expand_kernel(const float* __restrict__ tab,
                                   float* __restrict__ biasT) {
  int idx = blockIdx.x * blockDim.x + threadIdx.x;   // 65536 total
  int head = idx >> 12, m = (idx >> 6) & 63, n = idx & 63;
  int mh = m >> 3, mw = m & 7, nh = n >> 3, nw = n & 7;
  int rid = (mh - nh + 7) * 15 + (mw - nw + 7);
  biasT[idx] = tab[rid * 16 + head];
}

// ---------------------------------------------------------------------------
// WMMA bf16 GEMM: C[M,N] = A[M,K] @ Bt[N,K]^T, epilogue variants.
// KK is compile-time so the K loop unrolls and software-pipelines.
//  EPI 0: +bias -> bf16      1: +bias,gelu -> bf16     2: +bias -> f32
//  EPI 3: +bias, window-reverse row permute, +addsrc[token] -> f32
//  EPI 4: +bias, +addsrc[row] -> f32
// ---------------------------------------------------------------------------
template <int EPI, int KK>
__global__ __launch_bounds__(256)
void gemm_bf16_kernel(const __bf16* __restrict__ A, const __bf16* __restrict__ Bt,
                      const float* __restrict__ bias, int M, int N,
                      float* __restrict__ fout, __bf16* __restrict__ bout,
                      const float* __restrict__ addsrc) {
  int lane = threadIdx.x & 31, wave = threadIdx.x >> 5;
  int wm = wave & 1, wn = wave >> 1;
  int half = lane >> 4, r = lane & 15;
  long m0 = (long)blockIdx.x * 128 + wm * 64;
  long n0 = (long)blockIdx.y * 128 + wn * 32;

  f32x8 acc[4][2] = {};
  const __bf16* Ab = A + m0 * KK;
  const __bf16* Bb = Bt + n0 * KK;

#pragma unroll 2
  for (int k = 0; k < KK; k += 32) {
    bf16x16 af[4], bfr[2];
#pragma unroll
    for (int i = 0; i < 4; ++i)
      af[i] = load_fragA(Ab + (size_t)i * 16 * KK + k, KK, r, half);
#pragma unroll
    for (int j = 0; j < 2; ++j)
      bfr[j] = load_fragB(Bb + (size_t)j * 16 * KK + k, KK, r, half);
    if (k + 64 < KK) {   // pull next tile lines toward the WGP (global_prefetch_b8)
      __builtin_prefetch(Ab + (size_t)r * KK + k + 64, 0, 1);
      __builtin_prefetch(Bb + (size_t)r * KK + k + 64, 0, 1);
    }
#pragma unroll
    for (int i = 0; i < 4; ++i)
#pragma unroll
      for (int j = 0; j < 2; ++j) acc[i][j] = wmma_bf16(af[i], bfr[j], acc[i][j]);
  }

#pragma unroll
  for (int i = 0; i < 4; ++i)
#pragma unroll
    for (int j = 0; j < 2; ++j)
#pragma unroll
      for (int e = 0; e < 8; ++e) {
        long m = m0 + i * 16 + e + 8 * half;
        long n = n0 + j * 16 + r;
        float v = acc[i][j][e] + bias[n];
        if (EPI == 1) v = gelu_f(v);
        if (EPI == 0 || EPI == 1) {
          bout[m * N + n] = f2bf(v);
        } else if (EPI == 2) {
          fout[m * N + n] = v;
        } else if (EPI == 3) {
          long t = win_reverse_token(m);
          fout[t * N + n] = v + addsrc[t * N + n];
        } else {
          fout[m * N + n] = v + addsrc[m * N + n];
        }
      }
}

// ---------------------------------------------------------------------------
// Window attention: one wave handles (window, head, 16-row tile)
// S (16x64) via 4 WMMAs, softmax in-register (raw ds_bpermute reductions),
// P@V (16x32) via 4 WMMAs through an LDS re-fragmentation of P.
// ---------------------------------------------------------------------------
__global__ __launch_bounds__(256)
void attn_kernel(const __bf16* __restrict__ q, const __bf16* __restrict__ kk,
                 const __bf16* __restrict__ vv, const float* __restrict__ biasT,
                 __bf16* __restrict__ out) {
  __shared__ __bf16 ldsP[8][16 * 64];   // per-wave P staging (2 KB each)
  int lane = threadIdx.x & 31, wave = threadIdx.x >> 5;
  int half = lane >> 4, r = lane & 15;
  long task = (long)blockIdx.x * 8 + wave;
  int mt = (int)(task & 3);
  int head = (int)((task >> 2) & 15);
  long win = task >> 6;
  const float scale = 0.17677669529663687f;   // 32^-0.5

  // precomputed byte addresses for xor-1/2/4/8 lane gathers (stay in-wave)
  int ax1 = (lane ^ 1) << 2, ax2 = (lane ^ 2) << 2;
  int ax4 = (lane ^ 4) << 2, ax8 = (lane ^ 8) << 2;

  const __bf16* qb = q + (win * 64 + mt * 16) * 512 + head * 32;
  const __bf16* kb = kk + (win * 64) * 512 + head * 32;
  const __bf16* vb = vv + (win * 64) * 512 + head * 32;

  // ---- S = q @ k^T : 4 column tiles, K = hd = 32 (one WMMA each)
  f32x8 s[4] = {};
  {
    bf16x16 aq = load_fragA(qb, 512, r, half);
#pragma unroll
    for (int nt = 0; nt < 4; ++nt) {
      bf16x16 bk = load_fragB(kb + (size_t)nt * 16 * 512, 512, r, half);
      s[nt] = wmma_bf16(aq, bk, s[nt]);
    }
  }

  // ---- scale + rel-pos bias + softmax (rows live in e + lane-half)
  const float* bT = biasT + (size_t)head * 4096;
  float inv[8];
#pragma unroll
  for (int e = 0; e < 8; ++e) {
    int ml = mt * 16 + e + 8 * half;
    float mx = -3.0e38f;
#pragma unroll
    for (int nt = 0; nt < 4; ++nt) {
      int n = nt * 16 + r;
      float val = s[nt][e] * scale + bT[ml * 64 + n];
      s[nt][e] = val;
      mx = fmaxf(mx, val);
    }
    mx = fmaxf(mx, bperm_f32(ax1, mx));
    mx = fmaxf(mx, bperm_f32(ax2, mx));
    mx = fmaxf(mx, bperm_f32(ax4, mx));
    mx = fmaxf(mx, bperm_f32(ax8, mx));
    float sum = 0.f;
#pragma unroll
    for (int nt = 0; nt < 4; ++nt) {
      float p = __expf(s[nt][e] - mx);
      s[nt][e] = p; sum += p;
    }
    sum += bperm_f32(ax1, sum);
    sum += bperm_f32(ax2, sum);
    sum += bperm_f32(ax4, sum);
    sum += bperm_f32(ax8, sum);
    inv[e] = 1.0f / sum;
  }

  // ---- stage P (bf16) into LDS, re-fragment as A operand
  __bf16* P = &ldsP[wave][0];
#pragma unroll
  for (int e = 0; e < 8; ++e) {
    int ml = e + 8 * half;
#pragma unroll
    for (int nt = 0; nt < 4; ++nt) P[ml * 64 + nt * 16 + r] = f2bf(s[nt][e] * inv[e]);
  }
  __syncthreads();

  // ---- O = P @ V : K = 64 (two 32-steps), N = 32 (two tiles)
  f32x8 o[2] = {};
#pragma unroll
  for (int ks = 0; ks < 64; ks += 32) {
    bf16x16 ap = load_fragA(P + ks, 64, r, half);
#pragma unroll
    for (int nt = 0; nt < 2; ++nt) {
      bf16x16 bv;
      int kb0 = ks + half * 16;
#pragma unroll
      for (int j = 0; j < 16; ++j)
        bv[j] = vb[(size_t)(kb0 + j) * 512 + nt * 16 + r];
      o[nt] = wmma_bf16(ap, bv, o[nt]);
    }
  }

  __bf16* ob = out + (win * 64 + mt * 16) * 512 + head * 32;
#pragma unroll
  for (int nt = 0; nt < 2; ++nt)
#pragma unroll
    for (int e = 0; e < 8; ++e) {
      int ml = e + 8 * half;
      ob[(size_t)ml * 512 + nt * 16 + r] = f2bf(o[nt][e]);
    }
}

// ---------------------------------------------------------------------------
// Host orchestration
// ---------------------------------------------------------------------------
extern "C" void kernel_launch(void* const* d_in, const int* in_sizes, int n_in,
                              void* d_out, int out_size, void* d_ws, size_t ws_size,
                              hipStream_t stream) {
  const float* x       = (const float*)d_in[0];
  const float* feature = (const float*)d_in[1];
  const float* up_w1   = (const float*)d_in[2];
  const float* up_b1   = (const float*)d_in[3];
  const float* up_w2   = (const float*)d_in[4];
  const float* up_b2   = (const float*)d_in[5];
  const float* n1_g    = (const float*)d_in[6];
  const float* n1_b    = (const float*)d_in[7];
  const float* wq      = (const float*)d_in[8];
  const float* bq      = (const float*)d_in[9];
  const float* wk      = (const float*)d_in[10];
  const float* bk      = (const float*)d_in[11];
  const float* wv      = (const float*)d_in[12];
  const float* bv      = (const float*)d_in[13];
  const float* wp      = (const float*)d_in[14];
  const float* bp      = (const float*)d_in[15];
  const float* btab    = (const float*)d_in[16];
  const float* n2_g    = (const float*)d_in[17];
  const float* n2_b    = (const float*)d_in[18];
  const float* mlp_w1  = (const float*)d_in[19];
  const float* mlp_b1  = (const float*)d_in[20];
  const float* mlp_w2  = (const float*)d_in[21];
  const float* mlp_b2  = (const float*)d_in[22];

  const long M = 65536;            // total tokens (4 * 16384)
  const int  C = 512, HH = 2048;
  const size_t MiB = 1ull << 20;
  char* w = (char*)d_ws;

  __bf16* xb   = (__bf16*)(w + 0);             // 64 MiB (aliased by attn later)
  __bf16* attn = xb;
  __bf16* t0   = (__bf16*)(w + 64 * MiB);      // 64 MiB (aliased by yn later)
  __bf16* yn   = t0;
  float*  unary= (float*)(w + 128 * MiB);      // 128 MiB
  float*  fwin = (float*)(w + 256 * MiB);      // 128 MiB (aliased by hbuf later)
  __bf16* hbuf = (__bf16*)(w + 256 * MiB);     // 256 MiB
  __bf16* xn   = (__bf16*)(w + 384 * MiB);     // 64 MiB
  __bf16* qb_  = (__bf16*)(w + 448 * MiB);     // 64 MiB
  __bf16* kb_  = (__bf16*)(w + 512 * MiB);     // 64 MiB
  __bf16* vb_  = (__bf16*)(w + 576 * MiB);     // 64 MiB
  float*  y    = (float*)(w + 640 * MiB);      // 128 MiB
  char*   wts  = w + 768 * MiB;
  __bf16* wt_up1 = (__bf16*)(wts);
  __bf16* wt_up2 = wt_up1 + 512 * 512;
  __bf16* wt_q   = wt_up2 + 512 * 512;
  __bf16* wt_k   = wt_q   + 512 * 512;
  __bf16* wt_v   = wt_k   + 512 * 512;
  __bf16* wt_p   = wt_v   + 512 * 512;
  __bf16* wt_m1  = wt_p   + 512 * 512;         // 2048 x 512
  __bf16* wt_m2  = wt_m1  + 2048 * 512;        // 512 x 2048
  float*  biasT  = (float*)(wt_m2 + 2048 * 512);

  dim3 b256(256), bT(32, 8), b128(128);

  // 1. x -> bf16
  cvt_bf16_kernel<<<(unsigned)(M * C / 256), b256, 0, stream>>>(x, xb, M * C);

  // 2. weight transposes (fp32 KxN -> bf16 NxK)
  transpose_w_kernel<<<dim3(16, 16), bT, 0, stream>>>(up_w1, wt_up1, 512, 512);
  transpose_w_kernel<<<dim3(16, 16), bT, 0, stream>>>(up_w2, wt_up2, 512, 512);
  transpose_w_kernel<<<dim3(16, 16), bT, 0, stream>>>(wq, wt_q, 512, 512);
  transpose_w_kernel<<<dim3(16, 16), bT, 0, stream>>>(wk, wt_k, 512, 512);
  transpose_w_kernel<<<dim3(16, 16), bT, 0, stream>>>(wv, wt_v, 512, 512);
  transpose_w_kernel<<<dim3(16, 16), bT, 0, stream>>>(wp, wt_p, 512, 512);
  transpose_w_kernel<<<dim3(64, 16), bT, 0, stream>>>(mlp_w1, wt_m1, 512, 2048);
  transpose_w_kernel<<<dim3(16, 64), bT, 0, stream>>>(mlp_w2, wt_m2, 2048, 512);

  // 3. rel-pos bias expand
  bias_expand_kernel<<<256, b256, 0, stream>>>(btab, biasT);

  // 4-5. unary = gelu(x@up_w1+b1) @ up_w2 + b2
  gemm_bf16_kernel<1, 512><<<dim3(512, 4), b256, 0, stream>>>(
      xb, wt_up1, up_b1, (int)M, C, nullptr, t0, nullptr);
  gemm_bf16_kernel<2, 512><<<dim3(512, 4), b256, 0, stream>>>(
      t0, wt_up2, up_b2, (int)M, C, unary, nullptr, nullptr);

  // 6. window partition of feature
  partition_kernel<<<dim3(4, 16, 512), bT, 0, stream>>>(feature, fwin);

  // 7. LN1 -> bf16
  layernorm_kernel<<<(unsigned)M, b128, 0, stream>>>(fwin, n1_g, n1_b, xn);

  // 8. q, k, v projections
  gemm_bf16_kernel<0, 512><<<dim3(512, 4), b256, 0, stream>>>(
      xn, wt_q, bq, (int)M, C, nullptr, qb_, nullptr);
  gemm_bf16_kernel<0, 512><<<dim3(512, 4), b256, 0, stream>>>(
      xn, wt_k, bk, (int)M, C, nullptr, kb_, nullptr);
  gemm_bf16_kernel<0, 512><<<dim3(512, 4), b256, 0, stream>>>(
      xn, wt_v, bv, (int)M, C, nullptr, vb_, nullptr);

  // 9. window attention (1024 win * 16 heads * 4 row-tiles = 65536 wave tasks)
  attn_kernel<<<8192, b256, 0, stream>>>(qb_, kb_, vb_, biasT, attn);

  // 10. projection + window reverse + unary residual -> y (f32)
  gemm_bf16_kernel<3, 512><<<dim3(512, 4), b256, 0, stream>>>(
      attn, wt_p, bp, (int)M, C, y, nullptr, unary);

  // 11. LN2 -> bf16
  layernorm_kernel<<<(unsigned)M, b128, 0, stream>>>(y, n2_g, n2_b, yn);

  // 12-13. MLP with residual into d_out
  gemm_bf16_kernel<1, 512><<<dim3(512, 16), b256, 0, stream>>>(
      yn, wt_m1, mlp_b1, (int)M, HH, nullptr, hbuf, nullptr);
  gemm_bf16_kernel<4, 2048><<<dim3(512, 4), b256, 0, stream>>>(
      hbuf, wt_m2, mlp_b2, (int)M, C, (float*)d_out, nullptr, y);
}
